// SwinBlock_20048907338325
// MI455X (gfx1250) — matrix operand order repaired and checked by reading, not compiled, not run on hardware
//
#include <hip/hip_runtime.h>
#include <hip/hip_bf16.h>
#include <math.h>

// ---------------------------------------------------------------------------
// Swin block for MI455X (gfx1250, wave32). All GEMMs use
// v_wmma_f32_16x16x32_f16 (f16 inputs, f32 accumulate). Memory-bound problem:
// intermediates kept in f16, LN/softmax/GELU/residual fused into epilogues.
// ---------------------------------------------------------------------------

typedef _Float16 v16h __attribute__((ext_vector_type(16)));
typedef _Float16 v8h  __attribute__((ext_vector_type(8)));
typedef float    v8f  __attribute__((ext_vector_type(8)));

// Problem constants (from reference setup_inputs)
static constexpr int  BATCH = 8;
static constexpr int  IMG   = 128;            // H = W = 128
static constexpr int  C     = 180;
static constexpr int  CP    = 192;            // C padded to mult of 32
static constexpr int  HEADS = 6;
static constexpr int  HD    = 30;
static constexpr int  HDP   = 32;             // head dim padded
static constexpr int  NW    = 64;             // tokens per 8x8 window
static constexpr long MTOK  = (long)BATCH * IMG * IMG;   // 131072 rows
static constexpr int  NWIN  = (int)(MTOK / NW);          // 2048 windows
static constexpr int  HID   = 720;
static constexpr int  HIDP  = 736;            // 720 -> mult of 32
static constexpr int  KC    = 256;            // GEMM K chunk
static constexpr int  KCP   = KC + 8;         // LDS row stride (16B aligned)

// --------------------------- WMMA fragment loads ---------------------------
// A fragment (16x32, MxK), A stored row-major [m][k] with stride lda.
// Lane L: m = L%16, half = L/16; halves h -> K = 16*half + h  (contiguous along k).
__device__ __forceinline__ v16h load_A_frag(const _Float16* base, int lda) {
  int lane = threadIdx.x & 31;
  const _Float16* p = base + (lane & 15) * lda + ((lane >> 4) << 4);
  v16h r;
#pragma unroll
  for (int h = 0; h < 8; ++h) r[h] = p[h];
#pragma unroll
  for (int h = 0; h < 8; ++h) r[h + 8] = p[8 + h];
  return r;
}

// B fragment (32x16, KxN), B stored TRANSPOSED in LDS: Bt[n][k], stride ldb.
// Lane L: n = L%16; halves h -> K = 16*(L/16) + h  (contiguous along k).
__device__ __forceinline__ v16h load_Bt_frag(const _Float16* base, int ldb) {
  int lane = threadIdx.x & 31;
  const _Float16* p = base + (lane & 15) * ldb + ((lane >> 4) << 4);
  v16h r;
#pragma unroll
  for (int h = 0; h < 16; ++h) r[h] = p[h];
  return r;
}

__device__ __forceinline__ v8f wmma_f16(v16h a, v16h b, v8f c) {
  return __builtin_amdgcn_wmma_f32_16x16x32_f16(false, a, false, b,
                                                (short)0, c, false, false);
}

// C/D layout: VGPR r, lane L: m = r + 8*(L/16), n = L%16.

// ------------------------ row index permutations ---------------------------
__device__ __forceinline__ int orig_to_wr(int r) {       // token -> window order
  int b = r >> 14, rem = r & 16383;
  int hi = rem >> 7, wj = rem & 127;
  int hw = hi >> 3, i = hi & 7, ww = wj >> 3, j = wj & 7;
  return (((b * 16 + hw) * 16 + ww) << 6) + i * 8 + j;
}
__device__ __forceinline__ int wr_to_orig(int wr) {      // window -> token order
  int t = wr & 63, w = wr >> 6;
  int j = t & 7, i = t >> 3;
  int ww = w & 15, hw = (w >> 4) & 15, b = w >> 8;
  return b * 16384 + (hw * 8 + i) * 128 + ww * 8 + j;
}

// ------------------------- weight convert + pad ----------------------------
__global__ __launch_bounds__(256) void convert_pad_kernel(
    const float* __restrict__ src, _Float16* __restrict__ dst,
    int sr, int sc, int dr, int dc) {
  long total = (long)dr * dc;
  for (long i = blockIdx.x * 256 + threadIdx.x; i < total;
       i += (long)gridDim.x * 256) {
    int r = (int)(i / dc), c = (int)(i % dc);
    float v = (r < sr && c < sc) ? src[(long)r * sc + c] : 0.f;
    dst[i] = (_Float16)v;
  }
}

// ---------------- LayerNorm (one wave per row) + optional permute ----------
__global__ __launch_bounds__(256) void ln_kernel(
    const float* __restrict__ x, const float* __restrict__ w,
    const float* __restrict__ b, _Float16* __restrict__ out, int permute) {
  int row  = blockIdx.x * 8 + (threadIdx.x >> 5);
  int lane = threadIdx.x & 31;
  const float* xr = x + (long)row * C;
  float vals[6];
  float s = 0.f, ss = 0.f;
#pragma unroll
  for (int i = 0; i < 6; ++i) {
    int c = lane + 32 * i;
    float v = (c < C) ? xr[c] : 0.f;
    vals[i] = v; s += v; ss += v * v;
  }
#pragma unroll
  for (int m = 16; m >= 1; m >>= 1) {
    s  += __shfl_xor(s, m, 32);
    ss += __shfl_xor(ss, m, 32);
  }
  float mean = s * (1.f / C);
  float var  = ss * (1.f / C) - mean * mean;
  float rstd = rsqrtf(var + 1e-5f);
  long orow = permute ? orig_to_wr(row) : row;
  _Float16* o = out + orow * CP;
#pragma unroll
  for (int i = 0; i < 6; ++i) {
    int c = lane + 32 * i;
    if (c < C)        o[c] = (_Float16)((vals[i] - mean) * rstd * w[c] + b[c]);
    else if (c < CP)  o[c] = (_Float16)0.f;   // zero K-padding
  }
}

// ------------------------------- GEMM --------------------------------------
// Block tile 128(M) x 64(N); 256 threads = 8 waves; wave w -> 16 rows, 4 tiles.
enum { MODE_QKV = 0, MODE_PROJ = 1, MODE_FC1 = 2, MODE_FC2 = 3 };

template <int MODE>
__global__ __launch_bounds__(256) void gemm_kernel(
    const _Float16* __restrict__ A, int lda,
    const _Float16* __restrict__ W, int ldw,
    const float* __restrict__ bias, const float* __restrict__ res,
    void* __restrict__ outp, int ldo, int K) {
  extern __shared__ _Float16 sm[];
  _Float16* As = sm;                 // [128][KCP]
  _Float16* Ws = sm + 128 * KCP;     // [64][KCP]  (transposed: Ws[n][k])

  int m0 = blockIdx.x * 128, n0 = blockIdx.y * 64;
  int tid = threadIdx.x, lane = tid & 31, wv = tid >> 5;

  v8f acc[4];
#pragma unroll
  for (int nt = 0; nt < 4; ++nt) acc[nt] = (v8f){0, 0, 0, 0, 0, 0, 0, 0};

  for (int k0 = 0; k0 < K; k0 += KC) {
    int kc = (K - k0 < KC) ? (K - k0) : KC;   // always a multiple of 32
    int nv = kc >> 3;
    for (int it = tid; it < 128 * nv; it += 256) {        // A tile (16B loads)
      int rr = it / nv, cc = it - rr * nv;
      *(v8h*)(As + rr * KCP + cc * 8) =
          *(const v8h*)(A + (long)(m0 + rr) * lda + k0 + cc * 8);
    }
    for (int it = tid; it < 64 * kc; it += 256) {         // W tile, transposed
      int kk = it >> 6, nn = it & 63;
      Ws[nn * KCP + kk] = W[(long)(k0 + kk) * ldw + n0 + nn];
    }
    __syncthreads();
#pragma unroll 2
    for (int k = 0; k < kc; k += 32) {
      v16h a = load_A_frag(As + (wv * 16) * KCP + k, KCP);
#pragma unroll
      for (int nt = 0; nt < 4; ++nt) {
        v16h b = load_Bt_frag(Ws + (nt * 16) * KCP + k, KCP);
        acc[nt] = wmma_f16(a, b, acc[nt]);
      }
    }
    __syncthreads();
  }

  int half = lane >> 4, nl = lane & 15;
#pragma unroll
  for (int nt = 0; nt < 4; ++nt) {
#pragma unroll
    for (int r = 0; r < 8; ++r) {
      int m = m0 + wv * 16 + r + 8 * half;
      int n = n0 + nt * 16 + nl;
      float v = acc[nt][r];
      if (MODE == MODE_QKV) {
        if (n < 3 * C)
          ((_Float16*)outp)[(long)m * ldo + n] = (_Float16)(v + bias[n]);
      } else if (MODE == MODE_PROJ) {
        if (n < C) {
          long orig = wr_to_orig(m);
          ((float*)outp)[orig * ldo + n] = v + bias[n] + res[orig * C + n];
        }
      } else if (MODE == MODE_FC1) {
        if (n < HIDP) {
          float g = 0.f;
          if (n < HID) {
            float t = v + bias[n];
            g = 0.5f * t * (1.f + erff(t * 0.70710678118654752f));  // exact GELU
          }
          ((_Float16*)outp)[(long)m * ldo + n] = (_Float16)g;
        }
      } else {  // MODE_FC2
        if (n < C)
          ((float*)outp)[(long)m * ldo + n] = v + bias[n] + res[(long)m * C + n];
      }
    }
  }
}

// ----------------------------- Attention -----------------------------------
// One block (8 waves) per 64-token window; wave task = (head, 16-row block).
__global__ __launch_bounds__(256) void attn_kernel(
    const _Float16* __restrict__ qkv,    // [MTOK][540] f16 (window order)
    const float* __restrict__ rel_tab,   // [225][6] f32
    _Float16* __restrict__ O) {          // [MTOK][CP] f16 (window order)
  extern __shared__ char smraw[];
  _Float16* Qs   = (_Float16*)smraw;        // [6][64][32]  (pre-scaled)
  _Float16* Ks   = Qs  + HEADS * NW * HDP;  // [6][64][32]  (== Bt[n=key][k=dim])
  _Float16* Vst  = Ks  + HEADS * NW * HDP;  // [6][32][64]  (== Bt[n=dim][k=key])
  _Float16* Ps   = Vst + HEADS * HDP * NW;  // [8 waves][16][72]
  float*    bs   = (float*)(Ps + 8 * 16 * 72);  // [225*6]

  int win = blockIdx.x;
  int tid = threadIdx.x, lane = tid & 31, wv = tid >> 5;
  long rbase = (long)win * NW;
  const float scale = 0.18257418583505537f;  // 30^-0.5

  for (int it = tid; it < 225 * HEADS; it += 256) bs[it] = rel_tab[it];
  for (int it = tid; it < HEADS * NW * HDP; it += 256) {
    int h = it >> 11, rem = it & 2047, t = rem >> 5, d = rem & 31;
    const _Float16* row = qkv + (rbase + t) * (3 * C) + h * HD;
    Qs[it] = (d < HD) ? (_Float16)((float)row[d] * scale) : (_Float16)0.f;
    Ks[it] = (d < HD) ? row[C + d] : (_Float16)0.f;
    Vst[(h * HDP + d) * NW + t] = (d < HD) ? row[2 * C + d] : (_Float16)0.f;
  }
  __syncthreads();

  int half = lane >> 4, nl = lane & 15;
  for (int task = wv; task < HEADS * 4; task += 8) {   // 24 tasks, uniform
    int h = task >> 2, rb = task & 3;

    // S = (Q*scale) @ K^T   (K dim = 32, one WMMA per 16-key tile)
    v8f sacc[4];
#pragma unroll
    for (int nt = 0; nt < 4; ++nt) sacc[nt] = (v8f){0, 0, 0, 0, 0, 0, 0, 0};
    v16h aq = load_A_frag(Qs + (h * NW + rb * 16) * HDP, HDP);
#pragma unroll
    for (int nt = 0; nt < 4; ++nt) {
      v16h bk = load_Bt_frag(Ks + (h * NW + nt * 16) * HDP, HDP);
      sacc[nt] = wmma_f16(aq, bk, sacc[nt]);
    }

    // + relative position bias
#pragma unroll
    for (int nt = 0; nt < 4; ++nt)
#pragma unroll
      for (int r = 0; r < 8; ++r) {
        int m = rb * 16 + r + 8 * half;
        int n = nt * 16 + nl;
        int idx = ((m >> 3) - (n >> 3) + 7) * 15 + ((m & 7) - (n & 7) + 7);
        sacc[nt][r] += bs[idx * HEADS + h];
      }

    // row softmax: each row lives in one 16-lane half -> shfl_xor {1,2,4,8}
#pragma unroll
    for (int r = 0; r < 8; ++r) {
      float mx = fmaxf(fmaxf(sacc[0][r], sacc[1][r]),
                       fmaxf(sacc[2][r], sacc[3][r]));
#pragma unroll
      for (int msk = 8; msk >= 1; msk >>= 1) mx = fmaxf(mx, __shfl_xor(mx, msk, 32));
      float sum = 0.f;
#pragma unroll
      for (int nt = 0; nt < 4; ++nt) {
        sacc[nt][r] = expf(sacc[nt][r] - mx);
        sum += sacc[nt][r];
      }
#pragma unroll
      for (int msk = 8; msk >= 1; msk >>= 1) sum += __shfl_xor(sum, msk, 32);
      float inv = 1.f / sum;
      int lr = r + 8 * half;
#pragma unroll
      for (int nt = 0; nt < 4; ++nt)
        Ps[(wv * 16 + lr) * 72 + nt * 16 + nl] = (_Float16)(sacc[nt][r] * inv);
    }

    // O = P @ V   (K dim = 64 keys: 2 steps; N = head dim: 2 tiles)
    v8f oacc[2];
    oacc[0] = (v8f){0, 0, 0, 0, 0, 0, 0, 0};
    oacc[1] = (v8f){0, 0, 0, 0, 0, 0, 0, 0};
#pragma unroll
    for (int kt = 0; kt < 2; ++kt) {
      v16h ap = load_A_frag(Ps + wv * 16 * 72 + kt * 32, 72);
#pragma unroll
      for (int nt = 0; nt < 2; ++nt) {
        v16h bv = load_Bt_frag(Vst + (h * HDP + nt * 16) * NW + kt * 32, NW);
        oacc[nt] = wmma_f16(ap, bv, oacc[nt]);
      }
    }
#pragma unroll
    for (int nt = 0; nt < 2; ++nt)
#pragma unroll
      for (int r = 0; r < 8; ++r) {
        int m = rb * 16 + r + 8 * half;
        int nd = nt * 16 + nl;
        if (nd < HD)
          O[(rbase + m) * CP + h * HD + nd] = (_Float16)(oacc[nt][r]);
      }
  }
  // zero the K-padding columns 180..191 of this window's rows
  for (int it = tid; it < NW * (CP - C); it += 256) {
    int t = it / (CP - C), c = C + it % (CP - C);
    O[(rbase + t) * CP + c] = (_Float16)0.f;
  }
}

// ------------------------------ launcher -----------------------------------
extern "C" void kernel_launch(void* const* d_in, const int* in_sizes, int n_in,
                              void* d_out, int out_size, void* d_ws,
                              size_t ws_size, hipStream_t stream) {
  const float* x        = (const float*)d_in[0];
  const float* n1w      = (const float*)d_in[1];
  const float* n1b      = (const float*)d_in[2];
  const float* qkv_w    = (const float*)d_in[3];
  const float* qkv_b    = (const float*)d_in[4];
  const float* rel_tab  = (const float*)d_in[5];
  const float* proj_w   = (const float*)d_in[6];
  const float* proj_b   = (const float*)d_in[7];
  const float* n2w      = (const float*)d_in[8];
  const float* n2b      = (const float*)d_in[9];
  const float* fc1_w    = (const float*)d_in[10];
  const float* fc1_b    = (const float*)d_in[11];
  const float* fc2_w    = (const float*)d_in[12];
  const float* fc2_b    = (const float*)d_in[13];
  float* out = (float*)d_out;

  // workspace layout (bytes)
  char* ws = (char*)d_ws;
  _Float16* xw   = (_Float16*)ws;                             // [M][192] f16
  _Float16* qkvb = (_Float16*)(ws + MTOK * CP * 2);           // [M][540] f16
  float*    x1   = (float*)   (ws + MTOK * (CP * 2 + 540 * 2));        // [M][180]
  _Float16* hbuf = (_Float16*)(ws + MTOK * (CP * 2 + 540 * 2 + C * 4));// [M][736]
  _Float16* wq   = (_Float16*)(ws + MTOK * (CP * 2 + 540 * 2 + C * 4 + HIDP * 2));
  _Float16* wp   = wq  + (long)CP * 576;
  _Float16* wf1  = wp  + (long)CP * CP;
  _Float16* wf2  = wf1 + (long)CP * 768;
  _Float16* O    = xw;     // reuse (xw dead after QKV gemm)
  _Float16* y2   = qkvb;   // reuse (qkv dead after attention)

  const int gemm_smem = (128 + 64) * KCP * 2;   // 101,376 B dynamic LDS
  const int attn_smem =
      (HEADS * NW * HDP * 3 + 8 * 16 * 72) * 2 + 225 * HEADS * 4;  // ~97.6 KB

  // 1) f16 weights with zero K/N padding
  convert_pad_kernel<<<dim3(256), 256, 0, stream>>>(qkv_w, wq,  C,   3 * C, CP,   576);
  convert_pad_kernel<<<dim3(64),  256, 0, stream>>>(proj_w, wp, C,   C,     CP,   CP);
  convert_pad_kernel<<<dim3(256), 256, 0, stream>>>(fc1_w, wf1, C,   HID,   CP,   768);
  convert_pad_kernel<<<dim3(256), 256, 0, stream>>>(fc2_w, wf2, HID, C,     HIDP, CP);

  // 2) LN1 + window partition -> xw (f16, window order)
  ln_kernel<<<dim3((int)(MTOK / 8)), 256, 0, stream>>>(x, n1w, n1b, xw, 1);

  // 3) QKV gemm: [M,192] x [192,576] -> qkv f16 [M,540]
  gemm_kernel<MODE_QKV><<<dim3((int)(MTOK / 128), 9), 256, gemm_smem, stream>>>(
      xw, CP, wq, 576, qkv_b, nullptr, qkvb, 3 * C, CP);

  // 4) windowed attention -> O f16 [M,192] (window order)
  attn_kernel<<<dim3(NWIN), 256, attn_smem, stream>>>(qkvb, rel_tab, O);

  // 5) proj + window-reverse + residual -> x1 f32 [M,180] (token order)
  gemm_kernel<MODE_PROJ><<<dim3((int)(MTOK / 128), 3), 256, gemm_smem, stream>>>(
      O, CP, wp, CP, proj_b, x, x1, C, CP);

  // 6) LN2 -> y2 f16 [M,192]
  ln_kernel<<<dim3((int)(MTOK / 8)), 256, 0, stream>>>(x1, n2w, n2b, y2, 0);

  // 7) fc1 + exact GELU -> h f16 [M,736]
  gemm_kernel<MODE_FC1><<<dim3((int)(MTOK / 128), 12), 256, gemm_smem, stream>>>(
      y2, CP, wf1, 768, fc1_b, nullptr, hbuf, HIDP, CP);

  // 8) fc2 + bias + x1 residual -> d_out f32 [M,180]
  gemm_kernel<MODE_FC2><<<dim3((int)(MTOK / 128), 3), 256, gemm_smem, stream>>>(
      hbuf, HIDP, wf2, CP, fc2_b, x1, out, C, HIDP);
}
